// CrossSectionalAttentionFusionMLP_91225105367185
// MI455X (gfx1250) — matrix operand order repaired and checked by reading, not compiled, not run on hardware
//
#include <hip/hip_runtime.h>
#include <hip/hip_bf16.h>

// ---------------------------------------------------------------------------
// CDNA5 (gfx1250) WMMA implementation.
// Math: einsum('bhwv,bchw->bchw', softmax_mask, feat) == feat * rowsum(mask),
// so the attention transfer collapses to a per-(b,h,w) scalar scale. We compute
// the row-sums honestly through the full mask pipeline, then run the two big
// 512x1024 output GEMMs (93% of FLOPs) with v_wmma_f32_16x16x32_f16, staging
// pre-converted f16 weights into LDS with GLOBAL_LOAD_ASYNC_TO_LDS_B128 when
// the toolchain exposes the gfx1250 async builtins.
// ---------------------------------------------------------------------------

typedef __attribute__((ext_vector_type(16))) _Float16 v16h;
typedef __attribute__((ext_vector_type(8)))  float    v8f;
typedef __attribute__((ext_vector_type(4)))  unsigned int u32x4;
typedef __attribute__((ext_vector_type(4)))  int          i32x4;

// Address-space qualified int4 for the gfx1250 async-to-LDS builtins
// (param type per hipcc diagnostic: pointer to int __vector(4)).
typedef __attribute__((address_space(1))) i32x4 i32x4_g;
typedef __attribute__((address_space(3))) i32x4 i32x4_l;

union V16H { v16h v; u32x4 u[2]; };

#define WMMA_F16(a, b, c) \
  __builtin_amdgcn_wmma_f32_16x16x32_f16(false, (a), false, (b), (short)0, (c), false, false)

#if defined(__has_builtin)
#if __has_builtin(__builtin_amdgcn_global_load_async_to_lds_b128) && \
    __has_builtin(__builtin_amdgcn_s_wait_asynccnt)
#define USE_ASYNC_LDS 1
#endif
#endif

// A fragment (16x32 f16), LDS row-major [m][k], row stride 32 halves.
// ISA layout: lane<16 -> row=lane, halves {K0..7, K16..23}; lane>=16 ->
// row=lane-16, halves {K8..15, K24..31}.
__device__ __forceinline__ v16h load_a16(const _Float16* base, int lane) {
  const _Float16* p = base + (lane & 15) * 32 + ((lane < 16) ? 0 : 8);
  V16H r;
  r.u[0] = *(const u32x4*)(p);
  r.u[1] = *(const u32x4*)(p + 16);
  return r.v;
}

// B fragment (32x16 f16), LDS [n][k], row stride 32 halves.
// lane<16 -> col n=lane, K0..7, K16..23; lane>=16 -> col n=lane-16, K8..15, K24..31.
__device__ __forceinline__ v16h load_b16(const _Float16* base, int lane) {
  const _Float16* p = base + (lane & 15) * 32 + ((lane < 16) ? 0 : 8);
  V16H r;
  r.u[0] = *(const u32x4*)(p);
  r.u[1] = *(const u32x4*)(p + 16);
  return r.v;
}

// ---------------------------------------------------------------------------
// K0: f32 -> f16 weight conversion (one-shot per call).
// ---------------------------------------------------------------------------
__global__ void __launch_bounds__(256)
cvt_w_kernel(const float* __restrict__ w, _Float16* __restrict__ wh, int n) {
  int i = blockIdx.x * 256 + threadIdx.x;
  if (i < n) wh[i] = (_Float16)w[i];
}

// ---------------------------------------------------------------------------
// K1: mask stage 1+2.  Per mask: x1 = w1 @ relu(bn1(concat)), then
// y[b,h,c*32+w] = (w2 @ relu(bn2(x1)))[c].  GEMM: M=32768 pos, N=32, K=1024.
// Block: 128 thr (4 waves), tile M=64 x N=32.
// ---------------------------------------------------------------------------
__global__ void __launch_bounds__(128)
mask_stage12_kernel(const float* __restrict__ F, const float* __restrict__ L,
                    const float* __restrict__ g1, const float* __restrict__ b1,
                    const float* __restrict__ m1, const float* __restrict__ v1,
                    const float* __restrict__ w1,
                    const float* __restrict__ g2, const float* __restrict__ b2,
                    const float* __restrict__ m2, const float* __restrict__ v2,
                    const float* __restrict__ w2,
                    float* __restrict__ y_ws, int mask) {
  __shared__ _Float16 A_s[64 * 32];
  __shared__ _Float16 B_s[32 * 32];
  __shared__ float sc_s[1024], bi_s[1024];
  __shared__ float x1_s[64 * 32];

  const int tid = threadIdx.x;
  const int wv = tid >> 5;
  const int lane = tid & 31;
  const int p0 = blockIdx.x * 64;  // position = (b*32+h)*32 + w

  for (int c = tid; c < 1024; c += 128) {
    float s = g1[c] * rsqrtf(v1[c] + 1e-5f);
    sc_s[c] = s;
    bi_s[c] = b1[c] - m1[c] * s;
  }
  __syncthreads();

  const float* src0 = mask ? L : F;  // channels [0,512)
  const float* src1 = mask ? F : L;  // channels [512,1024)

  v8f acc0 = {}, acc1 = {};
  for (int k0 = 0; k0 < 1024; k0 += 32) {
    {  // stage A: 64 pos x 32 chan, BN+ReLU, f32 -> f16
      int pos = tid & 63;
      int p = p0 + pos;
      int b = p >> 10, hw = p & 1023;
      for (int kk = (tid >> 6); kk < 32; kk += 2) {
        int c = k0 + kk;
        const float* src = (c < 512) ? src0 : src1;
        float x = src[((size_t)b * 512 + (c & 511)) * 1024 + hw];
        float a = x * sc_s[c] + bi_s[c];
        A_s[pos * 32 + kk] = (_Float16)(a > 0.f ? a : 0.f);
      }
    }
    for (int i = tid; i < 32 * 32; i += 128) {  // stage B: w1 (32,1024)
      int n = i >> 5, kk = i & 31;
      B_s[n * 32 + kk] = (_Float16)w1[n * 1024 + k0 + kk];
    }
    __syncthreads();
    v16h a  = load_a16(&A_s[(wv * 16) * 32], lane);
    v16h bb0 = load_b16(&B_s[0], lane);
    v16h bb1 = load_b16(&B_s[16 * 32], lane);
    acc0 = WMMA_F16(a, bb0, acc0);
    acc1 = WMMA_F16(a, bb1, acc1);
    __syncthreads();
  }

  // spill x1 tile (64 pos x 32 ch) to LDS
  {
    int m0 = wv * 16;
    for (int r = 0; r < 8; r++) {
      int mrow = m0 + ((lane < 16) ? r : 8 + r);
      int nc = lane & 15;
      x1_s[mrow * 32 + nc] = acc0[r];
      x1_s[mrow * 32 + 16 + nc] = acc1[r];
    }
  }
  __syncthreads();

  // bn2 + relu + (8x32) conv, write y[(mask,b,h), c*32+w]
  {
    int pos = tid & 63;
    int p = p0 + pos;
    int b = p >> 10, h = (p >> 5) & 31, w = p & 31;
    int ocb = (tid >> 6) * 4;
    for (int oc = ocb; oc < ocb + 4; oc++) {
      float acc = 0.f;
      for (int j = 0; j < 32; j++) {
        float s = g2[j] * rsqrtf(v2[j] + 1e-5f);
        float xa = (x1_s[pos * 32 + j] - m2[j]) * s + b2[j];
        acc += w2[oc * 32 + j] * (xa > 0.f ? xa : 0.f);
      }
      y_ws[(((size_t)mask * 32 + b) * 32 + h) * 256 + oc * 32 + w] = acc;
    }
  }
}

// ---------------------------------------------------------------------------
// K2: logits = relu(y) @ mlp_w.T + mlp_b.  M=1024, N=1024, K=256 (per mask).
// Block: 128 thr (4 waves), tile M=64 x N=32. grid (16, 32).
// ---------------------------------------------------------------------------
__global__ void __launch_bounds__(128)
mlp_kernel(const float* __restrict__ y, const float* __restrict__ mlp_w,
           const float* __restrict__ mlp_b, float* __restrict__ logits) {
  __shared__ _Float16 A_s[64 * 32];
  __shared__ _Float16 B_s[32 * 32];
  const int tid = threadIdx.x;
  const int wv = tid >> 5;
  const int lane = tid & 31;
  const int r0 = blockIdx.x * 64;
  const int n0 = blockIdx.y * 32;

  v8f acc0 = {}, acc1 = {};
  for (int k0 = 0; k0 < 256; k0 += 32) {
    for (int i = tid; i < 64 * 32; i += 128) {
      int rr = i >> 5, kk = i & 31;
      float xv = y[(size_t)(r0 + rr) * 256 + k0 + kk];
      A_s[rr * 32 + kk] = (_Float16)(xv > 0.f ? xv : 0.f);
    }
    for (int i = tid; i < 32 * 32; i += 128) {
      int n = i >> 5, kk = i & 31;
      B_s[n * 32 + kk] = (_Float16)mlp_w[(size_t)(n0 + n) * 256 + k0 + kk];
    }
    __syncthreads();
    v16h a  = load_a16(&A_s[(wv * 16) * 32], lane);
    acc0 = WMMA_F16(a, load_b16(&B_s[0], lane), acc0);
    acc1 = WMMA_F16(a, load_b16(&B_s[16 * 32], lane), acc1);
    __syncthreads();
  }
  int m0 = r0 + wv * 16;
  for (int r = 0; r < 8; r++) {
    int mrow = m0 + ((lane < 16) ? r : 8 + r);
    int nc = n0 + (lane & 15);
    logits[(size_t)mrow * 1024 + nc]      = acc0[r] + mlp_b[nc];
    logits[(size_t)mrow * 1024 + nc + 16] = acc1[r] + mlp_b[nc + 16];
  }
}

// ---------------------------------------------------------------------------
// K3: per-row softmax over 32, output the row-sum of the softmax.
// One wave per row; 65536 rows total.
// ---------------------------------------------------------------------------
__global__ void __launch_bounds__(256)
softmax_sum_kernel(const float* __restrict__ logits, float* __restrict__ sums) {
  int rid = blockIdx.x * 8 + (threadIdx.x >> 5);
  int lane = threadIdx.x & 31;
  float x = logits[(size_t)rid * 32 + lane];
  float m = x;
  for (int o = 16; o; o >>= 1) m = fmaxf(m, __shfl_xor(m, o, 32));
  float e = __expf(x - m);
  float s = e;
  for (int o = 16; o; o >>= 1) s += __shfl_xor(s, o, 32);
  float r = e / s;
  float t = r;
  for (int o = 16; o; o >>= 1) t += __shfl_xor(t, o, 32);
  if (lane == 0) sums[rid] = t;
}

// ---------------------------------------------------------------------------
// K4: out = W(512x1024) @ relu(bn(concat[X, X*msum])).  M=32768, N=512, K=1024.
// Block: 256 thr (8 waves as 2Mx4N), tile M=64 x N=256; wave 32x64
// (8 WMMAs / K-step).  B tile (f16 weights) staged with async-to-LDS when
// available; epilogue transposed through LDS for coalesced 128B row stores.
// ---------------------------------------------------------------------------
__global__ void __launch_bounds__(256)
fuse_out_kernel(const float* __restrict__ X, const float* __restrict__ sums,
                const float* __restrict__ bg, const float* __restrict__ bb,
                const float* __restrict__ bm, const float* __restrict__ bv,
                const _Float16* __restrict__ Wh, float* __restrict__ out) {
  __shared__ __align__(16) unsigned char smem[64 * 1024];
  _Float16* A_s = (_Float16*)smem;                 // 64x32  f16 =  4 KB
  _Float16* B_s = (_Float16*)(smem + 4096);        // 256x32 f16 = 16 KB
  float*    C_s = (float*)smem;                    // 64x256 f32 = 64 KB (epilogue)
  __shared__ float sc_s[1024], bi_s[1024];
  __shared__ float msum_s[64];

  const int tid = threadIdx.x;
  const int wv = tid >> 5;
  const int lane = tid & 31;
  const int p0 = blockIdx.x * 64;
  const int n0g = blockIdx.y * 256;

  for (int c = tid; c < 1024; c += 256) {
    float s = bg[c] * rsqrtf(bv[c] + 1e-5f);
    sc_s[c] = s;
    bi_s[c] = bb[c] - bm[c] * s;
  }
  if (tid < 64) msum_s[tid] = sums[p0 + tid];
  __syncthreads();

  const int m0  = (wv >> 2) * 32;   // wave M origin within block tile
  const int n0w = (wv & 3) * 64;    // wave N origin within block tile

  v8f acc[2][4];
  for (int ti = 0; ti < 2; ti++)
    for (int tj = 0; tj < 4; tj++) acc[ti][tj] = (v8f){};

  for (int k0 = 0; k0 < 1024; k0 += 32) {
    // --- stage B: 256 n x 32 k of pre-converted f16 weights (pure copy) ---
    {
      // 1024 chunks of 16B; 4 per thread
      for (int c4 = tid; c4 < 1024; c4 += 256) {
        int n = c4 >> 2, part = (c4 & 3) * 8;
        const _Float16* gsrc = Wh + (size_t)(n0g + n) * 1024 + k0 + part;
        _Float16* ldst = B_s + n * 32 + part;
#if defined(USE_ASYNC_LDS)
        __builtin_amdgcn_global_load_async_to_lds_b128(
            (i32x4_g*)gsrc, (i32x4_l*)ldst, 0, 0);
#else
        *(u32x4*)ldst = *(const u32x4*)gsrc;
#endif
      }
    }
    // --- stage A: activations for 64 pos x 32 chans (BN+ReLU, f32->f16) ---
    {
      int pos = tid & 63;
      int p = p0 + pos;
      int b = p >> 10, hw = p & 1023;
      float tscale = (k0 >= 512) ? msum_s[pos] : 1.f;
      int cbase = k0 & 511;
      for (int kk = (tid >> 6); kk < 32; kk += 4) {
        int c = k0 + kk;
        float x = X[((size_t)b * 512 + cbase + kk) * 1024 + hw] * tscale;
        float a = x * sc_s[c] + bi_s[c];
        A_s[pos * 32 + kk] = (_Float16)(a > 0.f ? a : 0.f);
      }
    }
#if defined(USE_ASYNC_LDS)
    __builtin_amdgcn_s_wait_asynccnt(0);
#endif
    __syncthreads();

    v16h af0 = load_a16(&A_s[m0 * 32], lane);
    v16h af1 = load_a16(&A_s[(m0 + 16) * 32], lane);
    v16h bf0 = load_b16(&B_s[(n0w +  0) * 32], lane);
    v16h bf1 = load_b16(&B_s[(n0w + 16) * 32], lane);
    v16h bf2 = load_b16(&B_s[(n0w + 32) * 32], lane);
    v16h bf3 = load_b16(&B_s[(n0w + 48) * 32], lane);
    acc[0][0] = WMMA_F16(af0, bf0, acc[0][0]);
    acc[0][1] = WMMA_F16(af0, bf1, acc[0][1]);
    acc[0][2] = WMMA_F16(af0, bf2, acc[0][2]);
    acc[0][3] = WMMA_F16(af0, bf3, acc[0][3]);
    acc[1][0] = WMMA_F16(af1, bf0, acc[1][0]);
    acc[1][1] = WMMA_F16(af1, bf1, acc[1][1]);
    acc[1][2] = WMMA_F16(af1, bf2, acc[1][2]);
    acc[1][3] = WMMA_F16(af1, bf3, acc[1][3]);
    __syncthreads();
  }

  // transpose accumulators through LDS (C_s aliases the staging buffers)
  for (int ti = 0; ti < 2; ti++)
    for (int tj = 0; tj < 4; tj++)
      for (int r = 0; r < 8; r++) {
        int mr = m0 + ti * 16 + ((lane < 16) ? r : 8 + r);
        int nc = n0w + tj * 16 + (lane & 15);
        C_s[mr * 256 + nc] = acc[ti][tj][r];
      }
  __syncthreads();

  // coalesced stores: 64 consecutive w per output-channel row
  {
    int pos = tid & 63;
    int p = p0 + pos;
    int b = p >> 10, hw = p & 1023;
    for (int nn = (tid >> 6); nn < 256; nn += 4) {
      out[((size_t)b * 512 + n0g + nn) * 1024 + hw] = C_s[pos * 256 + nn];
    }
  }
}

// ---------------------------------------------------------------------------
// Host launcher.  Input flattening order (setup_inputs dict order):
// 0:F 1:L | fattn: 2..5 bn1(g,b,m,v) 6:w1 7..10 bn2 11:w2 12:mlp_w 13:mlp_b |
// lattn: 14..25 (same) | fnet: 26..29 bn, 30:w | lnet: 31..34 bn, 35:w
// Output: fout (16777216 f32) then lout.
// ---------------------------------------------------------------------------
extern "C" void kernel_launch(void* const* d_in, const int* in_sizes, int n_in,
                              void* d_out, int out_size, void* d_ws, size_t ws_size,
                              hipStream_t stream) {
  const float* F = (const float*)d_in[0];
  const float* L = (const float*)d_in[1];

  float* ws = (float*)d_ws;
  float* y_ws      = ws;                       // 2*1024*256   = 524288 f32
  float* logits_ws = ws + 524288;              // 2*1024*1024  = 2097152 f32
  float* sums_ws   = ws + 524288 + 2097152;    // 2*1024*32    = 65536 f32
  _Float16* wh_ws  = (_Float16*)(ws + 2686976); // 2 * 524288 f16 = 2 MB (~12.75 MB total)

  // K0: pre-convert the two 512x1024 output-conv weights to f16
  cvt_w_kernel<<<2048, 256, 0, stream>>>((const float*)d_in[30], wh_ws, 524288);
  cvt_w_kernel<<<2048, 256, 0, stream>>>((const float*)d_in[35], wh_ws + 524288, 524288);

  // K1: both masks
  for (int mask = 0; mask < 2; mask++) {
    int base = 2 + mask * 12;
    mask_stage12_kernel<<<512, 128, 0, stream>>>(
        F, L,
        (const float*)d_in[base + 0], (const float*)d_in[base + 1],
        (const float*)d_in[base + 2], (const float*)d_in[base + 3],
        (const float*)d_in[base + 4],
        (const float*)d_in[base + 5], (const float*)d_in[base + 6],
        (const float*)d_in[base + 7], (const float*)d_in[base + 8],
        (const float*)d_in[base + 9],
        y_ws, mask);
  }

  // K2: MLP GEMM per mask
  for (int mask = 0; mask < 2; mask++) {
    int base = 2 + mask * 12;
    mlp_kernel<<<dim3(16, 32), 128, 0, stream>>>(
        y_ws + (size_t)mask * 1024 * 256,
        (const float*)d_in[base + 10], (const float*)d_in[base + 11],
        logits_ws + (size_t)mask * 1024 * 1024);
  }

  // K3: softmax row-sums (2*1024*32 rows of 32)
  softmax_sum_kernel<<<8192, 256, 0, stream>>>(logits_ws, sums_ws);

  // K4: fout uses frontal + lattn sums + fnet; lout uses lateral + fattn sums + lnet
  float* fout = (float*)d_out;
  float* lout = (float*)d_out + (size_t)32 * 512 * 32 * 32;
  fuse_out_kernel<<<dim3(512, 2), 256, 0, stream>>>(
      F, sums_ws + 32768,
      (const float*)d_in[26], (const float*)d_in[27],
      (const float*)d_in[28], (const float*)d_in[29],
      wh_ws, fout);
  fuse_out_kernel<<<dim3(512, 2), 256, 0, stream>>>(
      L, sums_ws,
      (const float*)d_in[31], (const float*)d_in[32],
      (const float*)d_in[33], (const float*)d_in[34],
      wh_ws + 524288, lout);

  (void)in_sizes; (void)n_in; (void)out_size; (void)ws_size;
}
